// G2P_77945066487965
// MI455X (gfx1250) — matrix operand order repaired and checked by reading, not compiled, not run on hardware
//
#include <hip/hip_runtime.h>
#include <hip/hip_bf16.h>
#include <math.h>

typedef _Float16 half_t;
typedef __attribute__((ext_vector_type(8)))  _Float16 v8h;
typedef __attribute__((ext_vector_type(16))) _Float16 v16h;
typedef __attribute__((ext_vector_type(8)))  float    v8f;

#define DMODEL 512
#define BATCH  1024
#define TSRC   64
#define TTGT   64
#define NVOCAB 128

// ---------------------------------------------------------------------------
// f16 WMMA GEMM:  C[M,N] = act( A[M,K] @ B[K,N] + bias + Cadd )
//   A row-major f16 [M,K];  Bt = B-transposed row-major f16 [N,K].
//   Block tile 128x128, BK=32, 256 threads = 8 wave32 waves arranged 4(M)x2(N);
//   each wave computes a 32x64 sub-tile = 8 WMMA per K-chunk.
//   Double-buffered LDS staging: chunk k+1 is fetched global->regs while
//   chunk k is computed, then regs->LDS into the alternate buffer (one
//   barrier per iteration).  All fragments are loaded before the WMMA group
//   so a single s_wait_dscnt covers 8 back-to-back v_wmma issues.
// Requires: M%128==0, N%128==0, K%32==0, 16B-aligned pointers, lda/ldbt%8==0.
// ---------------------------------------------------------------------------
__global__ __launch_bounds__(256)
void gemm_wmma_f16(const half_t* __restrict__ A, int lda,
                   const half_t* __restrict__ Bt, int ldbt,
                   const float* __restrict__ Cadd, int ldca,
                   const float* __restrict__ bias,
                   float* __restrict__ Cout, int ldco,
                   half_t* __restrict__ Hout, int ldho,
                   int M, int N, int K, int act)
{
    __shared__ half_t As [2][128][40];  // 32 + 8 pad halfs; row stride 80B
    __shared__ half_t BsT[2][128][40];  // B^T tile: [n][k], K-fastest

    const int tid  = threadIdx.x;
    const int lane = tid & 31;
    const int wave = tid >> 5;
    const int hlf  = (lane >> 4) & 1;
    const int l16  = lane & 15;
    const int wm   = (wave & 3) * 32;     // wave M offset within block tile
    const int wn   = (wave >> 2) * 64;    // wave N offset within block tile
    const int row0 = blockIdx.y * 128;
    const int col0 = blockIdx.x * 128;

    v8f acc[2][4];
#pragma unroll
    for (int mt = 0; mt < 2; ++mt)
#pragma unroll
        for (int nt = 0; nt < 4; ++nt)
#pragma unroll
            for (int j = 0; j < 8; ++j) acc[mt][nt][j] = 0.0f;

    // staging: 512 slots of 8 halfs per matrix; thread handles slots tid, tid+256
    v8h ra[2], rb[2];
#pragma unroll
    for (int s = 0; s < 2; ++s) {
        int slot = tid + s * 256;
        int r = slot >> 2, c = (slot & 3) * 8;
        ra[s] = *(const v8h*)(A  + (size_t)(row0 + r) * lda  + c);
        rb[s] = *(const v8h*)(Bt + (size_t)(col0 + r) * ldbt + c);
    }
#pragma unroll
    for (int s = 0; s < 2; ++s) {
        int slot = tid + s * 256;
        int r = slot >> 2, c = (slot & 3) * 8;
        *(v8h*)&As [0][r][c] = ra[s];
        *(v8h*)&BsT[0][r][c] = rb[s];
    }
    __syncthreads();

    for (int kk = 0; kk < K; kk += 32) {
        const int  cur       = (kk >> 5) & 1;
        const bool have_next = (kk + 32) < K;

        // ---- prefetch next chunk into registers (overlaps with compute) ----
        if (have_next) {
#pragma unroll
            for (int s = 0; s < 2; ++s) {
                int slot = tid + s * 256;
                int r = slot >> 2, c = (slot & 3) * 8;
                ra[s] = *(const v8h*)(A  + (size_t)(row0 + r) * lda  + kk + 32 + c);
                rb[s] = *(const v8h*)(Bt + (size_t)(col0 + r) * ldbt + kk + 32 + c);
            }
        }

        // ---- load ALL fragments, then issue the 8 WMMAs back-to-back ----
        v16h af[2];
#pragma unroll
        for (int mt = 0; mt < 2; ++mt) {
            int am = wm + 16 * mt + l16;
            v8h alo = *(const v8h*)&As[cur][am][8 * hlf];
            v8h ahi = *(const v8h*)&As[cur][am][16 + 8 * hlf];
            af[mt] = __builtin_shufflevector(alo, ahi,
                0, 1, 2, 3, 4, 5, 6, 7, 8, 9, 10, 11, 12, 13, 14, 15);
        }
        v16h bf[4];
#pragma unroll
        for (int nt = 0; nt < 4; ++nt) {
            int bn = wn + 16 * nt + l16;
            v8h blo = *(const v8h*)&BsT[cur][bn][8 * hlf];
            v8h bhi = *(const v8h*)&BsT[cur][bn][16 + 8 * hlf];
            bf[nt] = __builtin_shufflevector(blo, bhi,
                0, 1, 2, 3, 4, 5, 6, 7, 8, 9, 10, 11, 12, 13, 14, 15);
        }
#pragma unroll
        for (int mt = 0; mt < 2; ++mt)
#pragma unroll
            for (int nt = 0; nt < 4; ++nt)
                acc[mt][nt] = __builtin_amdgcn_wmma_f32_16x16x32_f16(
                    false, af[mt], false, bf[nt], (short)0, acc[mt][nt],
                    false, false);

        // ---- commit prefetched chunk to the alternate LDS buffer ----
        if (have_next) {
            int nxt = cur ^ 1;
#pragma unroll
            for (int s = 0; s < 2; ++s) {
                int slot = tid + s * 256;
                int r = slot >> 2, c = (slot & 3) * 8;
                *(v8h*)&As [nxt][r][c] = ra[s];
                *(v8h*)&BsT[nxt][r][c] = rb[s];
            }
        }
        __syncthreads();
    }

    // ---- epilogue (C/D layout: VGPR r -> M = r + 8*half, N = lane&15) ----
#pragma unroll
    for (int mt = 0; mt < 2; ++mt) {
#pragma unroll
        for (int nt = 0; nt < 4; ++nt) {
#pragma unroll
            for (int r = 0; r < 8; ++r) {
                int row = row0 + wm + 16 * mt + 8 * hlf + r;
                int col = col0 + wn + 16 * nt + l16;
                float v = acc[mt][nt][r];
                if (bias) v += bias[col];
                if (Cadd) v += Cadd[(size_t)row * ldca + col];
                if (act == 1) v = tanhf(v);
                if (Cout) Cout[(size_t)row * ldco + col] = v;
                if (Hout) Hout[(size_t)row * ldho + col] = (half_t)v;
            }
        }
    }
}

// ---------------------------------------------------------------------------
// Small utility kernels
// ---------------------------------------------------------------------------
// dst (N x K, row-major f16) = transpose of src (K x N, row-major f32)
__global__ void cast_transpose_f16(half_t* __restrict__ dst,
                                   const float* __restrict__ src, int K, int N) {
    int i = blockIdx.x * blockDim.x + threadIdx.x;   // i = n*K + k (writes coalesced)
    if (i >= K * N) return;
    int n = i / K, k = i - n * K;
    dst[i] = (half_t)src[(size_t)k * N + n];
}
__global__ void add_vec(float* __restrict__ d, const float* __restrict__ a,
                        const float* __restrict__ b, int n) {
    int i = blockIdx.x * blockDim.x + threadIdx.x;
    if (i < n) d[i] = a[i] + b[i];
}
__global__ void fill_zero_f32(float* p, int n) {
    int i = blockIdx.x * blockDim.x + threadIdx.x;
    if (i < n) p[i] = 0.0f;
}
__global__ void fill_zero_f16(half_t* p, int n) {
    int i = blockIdx.x * blockDim.x + threadIdx.x;
    if (i < n) p[i] = (half_t)0.0f;
}
// out[row, 0:D) = f16(table[idx[row*stride], 0:D))
__global__ void gather_rows_f16(const int* __restrict__ idx, int idx_stride,
                                const float* __restrict__ tab, int d,
                                half_t* __restrict__ out, int ldo) {
    int row = blockIdx.x;
    int r = idx[(size_t)row * idx_stride];
    for (int c = threadIdx.x; c < d; c += blockDim.x)
        out[(size_t)row * ldo + c] = (half_t)tab[(size_t)r * d + c];
}

// ---------------------------------------------------------------------------
// Fused LSTM gate update.  G: [B, 4D] pre-activation (i|f|g|o), c: [B,D] f32.
// ---------------------------------------------------------------------------
__global__ void lstm_gates(const float* __restrict__ G, float* __restrict__ c,
                           half_t* __restrict__ h_out, int ldh,
                           half_t* __restrict__ seq_dst, int seq_ld,
                           half_t* __restrict__ hfin, half_t* __restrict__ cfin,
                           int ldfin) {
    int i = blockIdx.x * blockDim.x + threadIdx.x;
    if (i >= BATCH * DMODEL) return;
    int b = i / DMODEL, d = i % DMODEL;
    const float* g = G + (size_t)b * 4 * DMODEL;
    float gi = g[d], gf = g[DMODEL + d], gg = g[2 * DMODEL + d], go = g[3 * DMODEL + d];
    float si = 1.0f / (1.0f + __expf(-gi));
    float sf = 1.0f / (1.0f + __expf(-gf));
    float so = 1.0f / (1.0f + __expf(-go));
    float cn = sf * c[i] + si * tanhf(gg);
    float hn = so * tanhf(cn);
    c[i] = cn;
    h_out[(size_t)b * ldh + d] = (half_t)hn;
    if (seq_dst) seq_dst[(size_t)b * seq_ld + d] = (half_t)hn;
    if (hfin)    hfin[(size_t)b * ldfin + d] = (half_t)hn;
    if (cfin)    cfin[(size_t)b * ldfin + d] = (half_t)cn;
}

// ---------------------------------------------------------------------------
// Attention: one wave per (b, j) dot product  s[b,j] = scale * <q[b], K[b,j]>
// ---------------------------------------------------------------------------
__global__ __launch_bounds__(256)
void attn_scores(const float* __restrict__ q, const float* __restrict__ Km,
                 float* __restrict__ s, float scale) {
    int wave = threadIdx.x >> 5, lane = threadIdx.x & 31;
    int bj = blockIdx.x * 8 + wave;                 // 0 .. B*TS-1
    int b = bj / TSRC;
    const float* qp = q + (size_t)b * DMODEL;
    const float* kp = Km + (size_t)bj * DMODEL;
    float acc = 0.0f;
#pragma unroll
    for (int it = 0; it < DMODEL / 128; ++it) {
        int d = it * 128 + lane * 4;
        float4 qa = *(const float4*)(qp + d);
        float4 ka = *(const float4*)(kp + d);
        acc += qa.x * ka.x + qa.y * ka.y + qa.z * ka.z + qa.w * ka.w;
    }
#pragma unroll
    for (int m = 16; m >= 1; m >>= 1) acc += __shfl_xor(acc, m, 32);
    if (lane == 0) s[bj] = acc * scale;
}

// softmax over TS=64 per batch row; writes probs in-place and into attn output
__global__ __launch_bounds__(64)
void attn_softmax(float* __restrict__ s, float* __restrict__ attn_out /* +t*TS */) {
    __shared__ float sm[64];
    int b = blockIdx.x, j = threadIdx.x;
    float v = s[(size_t)b * TSRC + j];
    sm[j] = v; __syncthreads();
    for (int k = 32; k >= 1; k >>= 1) {
        if (j < k) sm[j] = fmaxf(sm[j], sm[j + k]);
        __syncthreads();
    }
    float mx = sm[0]; __syncthreads();
    float e = __expf(v - mx);
    sm[j] = e; __syncthreads();
    for (int k = 32; k >= 1; k >>= 1) {
        if (j < k) sm[j] += sm[j + k];
        __syncthreads();
    }
    float p = e / sm[0];
    s[(size_t)b * TSRC + j] = p;
    attn_out[(size_t)b * TTGT * TSRC + j] = p;
}

// ctx[b,d] = sum_j p[b,j] * V[b,j,d]  (written as f16 for the wo GEMM)
__global__ void attn_ctx(const float* __restrict__ s, const float* __restrict__ Vm,
                         half_t* __restrict__ ctx) {
    int i = blockIdx.x * blockDim.x + threadIdx.x;
    if (i >= BATCH * DMODEL) return;
    int b = i / DMODEL, d = i % DMODEL;
    const float* sp = s + (size_t)b * TSRC;
    const float* vp = Vm + (size_t)b * TSRC * DMODEL + d;
    float acc = 0.0f;
#pragma unroll 4
    for (int j = 0; j < TSRC; ++j) acc += sp[j] * vp[(size_t)j * DMODEL];
    ctx[i] = (half_t)acc;
}

// argmax over 128 logits per row, then gather next embedding into prev (f16)
__global__ __launch_bounds__(128)
void argmax_gather(const float* __restrict__ logits, int ldl,
                   const float* __restrict__ emb,
                   half_t* __restrict__ prev, int ldp) {
    __shared__ float sv[128];
    __shared__ int   si[128];
    int b = blockIdx.x, k = threadIdx.x;
    float v = logits[(size_t)b * ldl + k];
    sv[k] = v; si[k] = k; __syncthreads();
    for (int m = 64; m >= 1; m >>= 1) {
        if (k < m) {
            if (sv[k + m] > sv[k]) { sv[k] = sv[k + m]; si[k] = si[k + m]; }
        }
        __syncthreads();
    }
    int arg = si[0];
    for (int c = k; c < DMODEL; c += 128)
        prev[(size_t)b * ldp + c] = (half_t)emb[(size_t)arg * DMODEL + c];
}

// ---------------------------------------------------------------------------
// Host orchestration
// ---------------------------------------------------------------------------
extern "C" void kernel_launch(void* const* d_in, const int* in_sizes, int n_in,
                              void* d_out, int out_size, void* d_ws, size_t ws_size,
                              hipStream_t stream)
{
    const int*   src        = (const int*)d_in[0];
    const int*   tgt        = (const int*)d_in[1];
    const float* enc_emb_w  = (const float*)d_in[2];
    const float* enc_wi_f   = (const float*)d_in[3];
    const float* enc_wh_f   = (const float*)d_in[4];
    const float* enc_bi_f   = (const float*)d_in[5];
    const float* enc_bh_f   = (const float*)d_in[6];
    const float* enc_wi_b   = (const float*)d_in[7];
    const float* enc_wh_b   = (const float*)d_in[8];
    const float* enc_bi_b   = (const float*)d_in[9];
    const float* enc_bh_b   = (const float*)d_in[10];
    const float* enc_post_w = (const float*)d_in[11];
    const float* enc_post_b = (const float*)d_in[12];
    const float* h_post_w   = (const float*)d_in[13];
    const float* h_post_b   = (const float*)d_in[14];
    const float* c_post_w   = (const float*)d_in[15];
    const float* c_post_b   = (const float*)d_in[16];
    const float* wq_w       = (const float*)d_in[17];
    const float* wq_b       = (const float*)d_in[18];
    const float* wk_w       = (const float*)d_in[19];
    const float* wk_b       = (const float*)d_in[20];
    const float* wv_w       = (const float*)d_in[21];
    const float* wv_b       = (const float*)d_in[22];
    const float* wo_w       = (const float*)d_in[23];
    const float* wo_b       = (const float*)d_in[24];
    const float* dec_emb_w  = (const float*)d_in[25];
    const float* dec_wi     = (const float*)d_in[26];
    const float* dec_wh     = (const float*)d_in[27];
    const float* dec_bi     = (const float*)d_in[28];
    const float* dec_bh     = (const float*)d_in[29];
    const float* dec_post_w = (const float*)d_in[30];
    const float* dec_post_b = (const float*)d_in[31];

    float* attn_out = (float*)d_out;                                  // [B,TT,TS]
    float* res_out  = attn_out + (size_t)BATCH * TTGT * TSRC;         // [B,TT,128]

    // ---- workspace carving ----
    char* p = (char*)d_ws;
    auto carve = [&](size_t bytes) -> char* {
        char* r = p;
        p += (bytes + 255) & ~(size_t)255;
        return r;
    };
    // transposed f16 weights (N x K, row-major)
    half_t* wWiF     = (half_t*)carve(sizeof(half_t) * 512 * 2048);
    half_t* wWhF     = (half_t*)carve(sizeof(half_t) * 512 * 2048);
    half_t* wWiB     = (half_t*)carve(sizeof(half_t) * 512 * 2048);
    half_t* wWhB     = (half_t*)carve(sizeof(half_t) * 512 * 2048);
    half_t* wEncPost = (half_t*)carve(sizeof(half_t) * 1024 * 512);
    half_t* wHPost   = (half_t*)carve(sizeof(half_t) * 1024 * 512);
    half_t* wCPost   = (half_t*)carve(sizeof(half_t) * 1024 * 512);
    half_t* wQ       = (half_t*)carve(sizeof(half_t) * 512 * 512);
    half_t* wK       = (half_t*)carve(sizeof(half_t) * 512 * 512);
    half_t* wV       = (half_t*)carve(sizeof(half_t) * 512 * 512);
    half_t* wO       = (half_t*)carve(sizeof(half_t) * 512 * 512);
    half_t* wDecWi   = (half_t*)carve(sizeof(half_t) * 1024 * 2048);
    half_t* wDecWh   = (half_t*)carve(sizeof(half_t) * 512 * 2048);
    half_t* wDecPost = (half_t*)carve(sizeof(half_t) * 512 * 128);
    float*  encBF    = (float*)carve(sizeof(float) * 2048);
    float*  encBB    = (float*)carve(sizeof(float) * 2048);
    float*  decB     = (float*)carve(sizeof(float) * 2048);

    half_t* X       = (half_t*)carve(sizeof(half_t) * (size_t)BATCH * TSRC * DMODEL);
    half_t* seq     = (half_t*)carve(sizeof(half_t) * (size_t)BATCH * TSRC * 2 * DMODEL);
    half_t* encOutH = (half_t*)carve(sizeof(half_t) * (size_t)BATCH * TSRC * DMODEL);
    float*  Kmat    = (float*)carve(sizeof(float) * (size_t)BATCH * TSRC * DMODEL);
    float*  Vmat    = (float*)carve(sizeof(float) * (size_t)BATCH * TSRC * DMODEL);
    float*  G       = (float*)carve(sizeof(float) * (size_t)BATCH * 4 * DMODEL);
    half_t* hEnc    = (half_t*)carve(sizeof(half_t) * (size_t)BATCH * DMODEL);
    float*  cEnc    = (float*)carve(sizeof(float) * (size_t)BATCH * DMODEL);
    half_t* hcat    = (half_t*)carve(sizeof(half_t) * (size_t)BATCH * 2 * DMODEL);
    half_t* ccat    = (half_t*)carve(sizeof(half_t) * (size_t)BATCH * 2 * DMODEL);
    half_t* hDec    = (half_t*)carve(sizeof(half_t) * (size_t)BATCH * DMODEL);
    float*  cDec    = (float*)carve(sizeof(float) * (size_t)BATCH * DMODEL);
    half_t* xcat    = (half_t*)carve(sizeof(half_t) * (size_t)BATCH * 2 * DMODEL);
    float*  qbuf    = (float*)carve(sizeof(float) * (size_t)BATCH * DMODEL);
    float*  scores  = (float*)carve(sizeof(float) * (size_t)BATCH * TSRC);
    half_t* ctxH    = (half_t*)carve(sizeof(half_t) * (size_t)BATCH * DMODEL);

    auto cvtT = [&](half_t* dst, const float* s_, int K, int N) {
        int n = K * N;
        cast_transpose_f16<<<(n + 255) / 256, 256, 0, stream>>>(dst, s_, K, N);
    };
    auto gemm = [&](const half_t* A, int lda, const half_t* Bt, int ldbt,
                    const float* Cadd, int ldca, const float* bias,
                    float* Cout, int ldco, half_t* Hout, int ldho,
                    int M, int N, int K, int act) {
        dim3 g(N / 128, M / 128);
        gemm_wmma_f16<<<g, 256, 0, stream>>>(A, lda, Bt, ldbt, Cadd, ldca, bias,
                                             Cout, ldco, Hout, ldho, M, N, K, act);
    };

    // ---- one-time prep (per call, deterministic): transposed f16 weights ----
    cvtT(wWiF, enc_wi_f, 512, 2048);   cvtT(wWhF, enc_wh_f, 512, 2048);
    cvtT(wWiB, enc_wi_b, 512, 2048);   cvtT(wWhB, enc_wh_b, 512, 2048);
    cvtT(wEncPost, enc_post_w, 1024, 512);
    cvtT(wHPost, h_post_w, 1024, 512); cvtT(wCPost, c_post_w, 1024, 512);
    cvtT(wQ, wq_w, 512, 512); cvtT(wK, wk_w, 512, 512);
    cvtT(wV, wv_w, 512, 512); cvtT(wO, wo_w, 512, 512);
    cvtT(wDecWi, dec_wi, 1024, 2048);  cvtT(wDecWh, dec_wh, 512, 2048);
    cvtT(wDecPost, dec_post_w, 512, 128);
    add_vec<<<8, 256, 0, stream>>>(encBF, enc_bi_f, enc_bh_f, 2048);
    add_vec<<<8, 256, 0, stream>>>(encBB, enc_bi_b, enc_bh_b, 2048);
    add_vec<<<8, 256, 0, stream>>>(decB, dec_bi, dec_bh, 2048);

    gather_rows_f16<<<BATCH * TSRC, 256, 0, stream>>>(src, 1, enc_emb_w, DMODEL, X, DMODEL);

    const int nBD = BATCH * DMODEL;
    // ---- encoder forward ----
    fill_zero_f16<<<(nBD + 255) / 256, 256, 0, stream>>>(hEnc, nBD);
    fill_zero_f32<<<(nBD + 255) / 256, 256, 0, stream>>>(cEnc, nBD);
    for (int t = 0; t < TSRC; ++t) {
        gemm(X + (size_t)t * DMODEL, TSRC * DMODEL, wWiF, DMODEL,
             nullptr, 0, encBF, G, 4 * DMODEL, nullptr, 0,
             BATCH, 4 * DMODEL, DMODEL, 0);
        gemm(hEnc, DMODEL, wWhF, DMODEL,
             G, 4 * DMODEL, nullptr, G, 4 * DMODEL, nullptr, 0,
             BATCH, 4 * DMODEL, DMODEL, 0);
        bool last = (t == TSRC - 1);
        lstm_gates<<<(nBD + 255) / 256, 256, 0, stream>>>(
            G, cEnc, hEnc, DMODEL,
            seq + (size_t)t * 2 * DMODEL, TSRC * 2 * DMODEL,
            last ? hcat : nullptr, last ? ccat : nullptr, 2 * DMODEL);
    }
    // ---- encoder backward ----
    fill_zero_f16<<<(nBD + 255) / 256, 256, 0, stream>>>(hEnc, nBD);
    fill_zero_f32<<<(nBD + 255) / 256, 256, 0, stream>>>(cEnc, nBD);
    for (int t = 0; t < TSRC; ++t) {
        int ts = TSRC - 1 - t;
        gemm(X + (size_t)ts * DMODEL, TSRC * DMODEL, wWiB, DMODEL,
             nullptr, 0, encBB, G, 4 * DMODEL, nullptr, 0,
             BATCH, 4 * DMODEL, DMODEL, 0);
        gemm(hEnc, DMODEL, wWhB, DMODEL,
             G, 4 * DMODEL, nullptr, G, 4 * DMODEL, nullptr, 0,
             BATCH, 4 * DMODEL, DMODEL, 0);
        bool last = (t == TSRC - 1);
        lstm_gates<<<(nBD + 255) / 256, 256, 0, stream>>>(
            G, cEnc, hEnc, DMODEL,
            seq + (size_t)ts * 2 * DMODEL + DMODEL, TSRC * 2 * DMODEL,
            last ? hcat + DMODEL : nullptr, last ? ccat + DMODEL : nullptr, 2 * DMODEL);
    }

    // ---- post projections (bulk GEMMs) ----
    gemm(seq, 2 * DMODEL, wEncPost, 2 * DMODEL, nullptr, 0, enc_post_b,
         nullptr, 0, encOutH, DMODEL, BATCH * TSRC, DMODEL, 2 * DMODEL, 0);
    gemm(hcat, 2 * DMODEL, wHPost, 2 * DMODEL, nullptr, 0, h_post_b,
         nullptr, 0, hDec, DMODEL, BATCH, DMODEL, 2 * DMODEL, 1);
    gemm(ccat, 2 * DMODEL, wCPost, 2 * DMODEL, nullptr, 0, c_post_b,
         cDec, DMODEL, nullptr, 0, BATCH, DMODEL, 2 * DMODEL, 1);
    gemm(encOutH, DMODEL, wK, DMODEL, nullptr, 0, wk_b,
         Kmat, DMODEL, nullptr, 0, BATCH * TSRC, DMODEL, DMODEL, 0);
    gemm(encOutH, DMODEL, wV, DMODEL, nullptr, 0, wv_b,
         Vmat, DMODEL, nullptr, 0, BATCH * TSRC, DMODEL, DMODEL, 0);

    gather_rows_f16<<<BATCH, 256, 0, stream>>>(tgt, TTGT, dec_emb_w, DMODEL,
                                               xcat, 2 * DMODEL);

    // ---- decoder ----
    const float scale = 1.0f / sqrtf((float)DMODEL);
    for (int t = 0; t < TTGT; ++t) {
        gemm(xcat, 2 * DMODEL, wQ, DMODEL, nullptr, 0, wq_b,
             qbuf, DMODEL, nullptr, 0, BATCH, DMODEL, DMODEL, 0);
        attn_scores<<<BATCH * TSRC / 8, 256, 0, stream>>>(qbuf, Kmat, scores, scale);
        attn_softmax<<<BATCH, 64, 0, stream>>>(scores, attn_out + (size_t)t * TSRC);
        attn_ctx<<<(nBD + 255) / 256, 256, 0, stream>>>(scores, Vmat, ctxH);
        gemm(ctxH, DMODEL, wO, DMODEL, nullptr, 0, wo_b,
             nullptr, 0, xcat + DMODEL, 2 * DMODEL, BATCH, DMODEL, DMODEL, 0);
        gemm(xcat, 2 * DMODEL, wDecWi, 2 * DMODEL, nullptr, 0, decB,
             G, 4 * DMODEL, nullptr, 0, BATCH, 4 * DMODEL, 2 * DMODEL, 0);
        gemm(hDec, DMODEL, wDecWh, DMODEL, G, 4 * DMODEL, nullptr,
             G, 4 * DMODEL, nullptr, 0, BATCH, 4 * DMODEL, DMODEL, 0);
        lstm_gates<<<(nBD + 255) / 256, 256, 0, stream>>>(
            G, cDec, hDec, DMODEL, nullptr, 0, nullptr, nullptr, 0);
        gemm(hDec, DMODEL, wDecPost, DMODEL, nullptr, 0, dec_post_b,
             res_out + (size_t)t * NVOCAB, TTGT * NVOCAB, nullptr, 0,
             BATCH, NVOCAB, DMODEL, 0);
        argmax_gather<<<BATCH, 128, 0, stream>>>(res_out + (size_t)t * NVOCAB,
                                                 TTGT * NVOCAB, dec_emb_w,
                                                 xcat, 2 * DMODEL);
    }
}
